// GraphAttentionLayer_48421461295779
// MI455X (gfx1250) — matrix operand order repaired and checked by reading, not compiled, not run on hardware
//
#include <hip/hip_runtime.h>

typedef __attribute__((ext_vector_type(16))) _Float16 v16h;
typedef __attribute__((ext_vector_type(8)))  float    v8f;
typedef __attribute__((ext_vector_type(4)))  float    v4f;
typedef __attribute__((ext_vector_type(4)))  int      v4i;

#define N_NODES 8192
#define F_IN    256
#define F_OUT   64
#define GAT_ALPHA 0.2f
#define L2E 1.4426950408889634f

// ---------------------------------------------------------------------------
// Kernel 1: Wh = h @ W   (8192x256 @ 256x64), also emit f16 transposed copy
// WhT16[n][row] for contiguous WMMA B-operand loads in the main kernel.
// ---------------------------------------------------------------------------
__global__ __launch_bounds__(256) void k_wh(const float* __restrict__ h,
                                            const float* __restrict__ W,
                                            float* __restrict__ Wh,
                                            _Float16* __restrict__ WhT16) {
  const int tid = threadIdx.x;
  const int n   = tid & 63;                    // output feature
  const int row = blockIdx.x * 4 + (tid >> 6); // 4 rows per block
  const float* hrow = h + (size_t)row * F_IN;
  float acc = 0.f;
#pragma unroll 4
  for (int k = 0; k < F_IN; ++k)
    acc = fmaf(hrow[k], W[k * F_OUT + n], acc);
  Wh[(size_t)row * F_OUT + n]      = acc;
  WhT16[(size_t)n * N_NODES + row] = (_Float16)acc;
}

// ---------------------------------------------------------------------------
// Kernel 2: f1 = Wh @ a[:64], f2 = Wh @ a[64:]
// ---------------------------------------------------------------------------
__global__ __launch_bounds__(256) void k_f(const float* __restrict__ Wh,
                                           const float* __restrict__ a,
                                           float* __restrict__ f1,
                                           float* __restrict__ f2) {
  const int i = blockIdx.x * 256 + threadIdx.x;
  const float* wr = Wh + (size_t)i * F_OUT;
  float s1 = 0.f, s2 = 0.f;
#pragma unroll 8
  for (int n = 0; n < F_OUT; ++n) {
    float v = wr[n];
    s1 = fmaf(v, a[n], s1);
    s2 = fmaf(v, a[F_OUT + n], s2);
  }
  f1[i] = s1;
  f2[i] = s2;
}

// ---------------------------------------------------------------------------
// Kernel 3: global max of f2 (single block reduce) -> softmax shift base
// ---------------------------------------------------------------------------
__global__ __launch_bounds__(256) void k_gmax(const float* __restrict__ f2,
                                              float* __restrict__ gmax) {
  __shared__ float red[256];
  float m = -1e30f;
  for (int i = threadIdx.x; i < N_NODES; i += 256) m = fmaxf(m, f2[i]);
  red[threadIdx.x] = m;
  __syncthreads();
  for (int s = 128; s > 0; s >>= 1) {
    if (threadIdx.x < s)
      red[threadIdx.x] = fmaxf(red[threadIdx.x], red[threadIdx.x + s]);
    __syncthreads();
  }
  if (threadIdx.x == 0) *gmax = red[0];
}

// ---------------------------------------------------------------------------
// Branchless: 4 probabilities packed into A-operand (f16) slots.
// nm2 = -m_row * log2(e). Masked entries get arg = -128 -> exp2 underflows,
// and the f16 convert flushes any residual denorm to exactly 0.
// ---------------------------------------------------------------------------
__device__ __forceinline__ void pgroup(v16h& a, int base, v4i Ai, v4f Fi,
                                       float f1r, float nm2) {
#pragma unroll
  for (int e = 0; e < 4; ++e) {
    float t   = f1r + Fi[e];
    float ee  = fmaxf(t, GAT_ALPHA * t);       // leaky_relu, branchless
    float arg = fmaf(ee, L2E, nm2);            // log2-domain shifted exponent
    arg       = (Ai[e] > 0) ? arg : -128.0f;   // mask via cndmask on the arg
    a[base + e] = (_Float16)__builtin_amdgcn_exp2f(arg); // single v_exp_f32
  }
}

// ---------------------------------------------------------------------------
// Main fused kernel: one wave owns 16 rows; single pass over all 8192
// columns in K=32 steps; P built in-register in WMMA f16 A-layout.
// 4 WMMAs accumulate P @ Wh (16x64, C-layout) and a 5th WMMA against an
// all-ones B tile accumulates the softmax row sums in the *same* C-layout,
// so normalization is elementwise (no shuffles).
// All global loads are issued at the top of the iteration so the ~100-op
// exp chain hides the B-tile (L2-hit) latency in-wave.
// ---------------------------------------------------------------------------
__global__ __launch_bounds__(128) void k_main(const int* __restrict__ adj,
                                              const float* __restrict__ f1,
                                              const float* __restrict__ f2,
                                              const float* __restrict__ gmaxp,
                                              const _Float16* __restrict__ WhT16,
                                              float* __restrict__ out) {
  const int  tid     = threadIdx.x;
  const int  wave    = tid >> 5;
  const int  lane    = tid & 31;
  const int  r       = lane & 15;     // A-layout: lane&15 == matrix row M
  const bool hi      = lane >= 16;    // lane-half selects K sub-ranges
  const int  rowbase = (blockIdx.x * 4 + wave) * 16;
  const int  row     = rowbase + r;

  const float f1r  = f1[row];
  const float gmax = *gmaxp;
  const float t0   = f1r + gmax;
  const float mrow = fmaxf(t0, GAT_ALPHA * t0); // per-row softmax shift
  const float nm2  = -mrow * L2E;

  // A-operand K offsets per lane-half: lo {0,4,8,12}, hi {16,20,24,28}
  const int kb  = hi ? 16 : 0;
  const int ko0 = kb, ko1 = kb + 4, ko2 = kb + 8, ko3 = kb + 12;

  const int* adjr = adj + (size_t)row * N_NODES;
  // B-operand: lane holds column n, K rows contiguous; half-offset 16 halves
  const _Float16* bb  = WhT16 + (hi ? 16 : 0);
  const _Float16* b0p = bb + (size_t)(r)      * N_NODES;
  const _Float16* b1p = bb + (size_t)(16 + r) * N_NODES;
  const _Float16* b2p = bb + (size_t)(32 + r) * N_NODES;
  const _Float16* b3p = bb + (size_t)(48 + r) * N_NODES;

  v16h bones;
#pragma unroll
  for (int i = 0; i < 16; ++i) bones[i] = (_Float16)1.0f;

  v8f acc[4] = {v8f{}, v8f{}, v8f{}, v8f{}};
  v8f accs   = {};   // row-sum tile (every N column holds the row sum)

  for (int j0 = 0; j0 < N_NODES; j0 += 32) {
    // prefetch adjacency 4KB ahead within this row
    int pj = j0 + 1024;
    if (pj < N_NODES) __builtin_prefetch(adjr + pj, 0, 3);

    // ---- issue ALL loads up front (adj + f2 + 4 B tiles) ----
    v4i A0 = *(const v4i*)(adjr + j0 + ko0);
    v4i A1 = *(const v4i*)(adjr + j0 + ko1);
    v4i A2 = *(const v4i*)(adjr + j0 + ko2);
    v4i A3 = *(const v4i*)(adjr + j0 + ko3);
    v4f F0 = *(const v4f*)(f2 + j0 + ko0);
    v4f F1 = *(const v4f*)(f2 + j0 + ko1);
    v4f F2 = *(const v4f*)(f2 + j0 + ko2);
    v4f F3 = *(const v4f*)(f2 + j0 + ko3);
    v16h b0 = *(const v16h*)(b0p + j0);
    v16h b1 = *(const v16h*)(b1p + j0);
    v16h b2 = *(const v16h*)(b2p + j0);
    v16h b3 = *(const v16h*)(b3p + j0);

    // ---- probability tile (covers B-load latency) ----
    v16h a;
    pgroup(a, 0,  A0, F0, f1r, nm2);
    pgroup(a, 4,  A1, F1, f1r, nm2);
    pgroup(a, 8,  A2, F2, f1r, nm2);
    pgroup(a, 12, A3, F3, f1r, nm2);

    acc[0] = __builtin_amdgcn_wmma_f32_16x16x32_f16(false, a, false, b0,
                                                    (short)0, acc[0], false, false);
    acc[1] = __builtin_amdgcn_wmma_f32_16x16x32_f16(false, a, false, b1,
                                                    (short)0, acc[1], false, false);
    acc[2] = __builtin_amdgcn_wmma_f32_16x16x32_f16(false, a, false, b2,
                                                    (short)0, acc[2], false, false);
    acc[3] = __builtin_amdgcn_wmma_f32_16x16x32_f16(false, a, false, b3,
                                                    (short)0, acc[3], false, false);
    accs   = __builtin_amdgcn_wmma_f32_16x16x32_f16(false, a, false, bones,
                                                    (short)0, accs, false, false);
  }

  // accs[k] already holds the row sum for exactly the row that acc[t][k]
  // belongs to in this lane (identical C-layout) -> elementwise normalize.
  const int mbase = rowbase + (hi ? 8 : 0);
#pragma unroll
  for (int k = 0; k < 8; ++k) {
    const float rcpl = __builtin_amdgcn_rcpf(accs[k]); // single v_rcp_f32
#pragma unroll
    for (int t = 0; t < 4; ++t) {
      float v = acc[t][k] * rcpl;
      // elu(alpha=1), branchless: exp2(v*log2e)-1 on the negative side
      float en = __builtin_amdgcn_exp2f(v * L2E) - 1.0f;
      float o  = v > 0.f ? v : en;
      out[(size_t)(mbase + k) * F_OUT + t * 16 + r] = o;
    }
  }
}

// ---------------------------------------------------------------------------
extern "C" void kernel_launch(void* const* d_in, const int* in_sizes, int n_in,
                              void* d_out, int out_size, void* d_ws, size_t ws_size,
                              hipStream_t stream) {
  const float* h   = (const float*)d_in[0];
  const int*   adj = (const int*)d_in[1];
  const float* W   = (const float*)d_in[2];
  const float* a   = (const float*)d_in[3];
  float*       out = (float*)d_out;

  char* ws = (char*)d_ws;
  float*    Wh    = (float*)ws;                                    // 2 MB
  _Float16* WhT16 = (_Float16*)(ws + (size_t)N_NODES * F_OUT * 4); // 1 MB
  float*    f1    = (float*)(ws + (size_t)3 * 1024 * 1024);
  float*    f2    = f1 + N_NODES;
  float*    gmax  = f2 + N_NODES;

  k_wh  <<<N_NODES / 4,   256, 0, stream>>>(h, W, Wh, WhT16);
  k_f   <<<N_NODES / 256, 256, 0, stream>>>(Wh, a, f1, f2);
  k_gmax<<<1,             256, 0, stream>>>(f2, gmax);
  k_main<<<N_NODES / 64,  128, 0, stream>>>(adj, f1, f2, gmax, WhT16, out);
}
